// GAT_29953101922534
// MI455X (gfx1250) — matrix operand order repaired and checked by reading, not compiled, not run on hardware
//
#include <hip/hip_runtime.h>
#include <hip/hip_bf16.h>

typedef __attribute__((ext_vector_type(16))) __bf16 v16bf;
typedef __attribute__((ext_vector_type(8)))  float  v8f;

#define N_NODES  20000
#define M_PAD    20032   // N_NODES rounded up to 64 for the 64x128 GEMM workgroup tile
#define N_EDGES  320000
#define N_GRAPHS 128
#define TPB 256

// ---------- helpers ----------
__device__ __forceinline__ unsigned short f2bf(float f) {
    unsigned u = __float_as_uint(f);
    u += 0x7FFFu + ((u >> 16) & 1u);      // round-to-nearest-even
    return (unsigned short)(u >> 16);
}
// monotone float<->uint order encoding (for atomicMax on floats incl. negatives)
__device__ __forceinline__ unsigned fenc(float f) {
    unsigned u = __float_as_uint(f);
    return (u & 0x80000000u) ? ~u : (u | 0x80000000u);
}
__device__ __forceinline__ float fdec(unsigned e) {
    unsigned u = (e & 0x80000000u) ? (e & 0x7FFFFFFFu) : ~e;
    return __uint_as_float(u);
}
__device__ __forceinline__ float lrelu(float v) { return v > 0.0f ? v : 0.2f * v; }

// ---------- elementwise / setup kernels ----------
__global__ void k_cast_bf16(const float* __restrict__ in, unsigned short* __restrict__ out, int n) {
    int i = blockIdx.x * blockDim.x + threadIdx.x;
    if (i < n) out[i] = f2bf(in[i]);
}
// in: [K,N] row-major f32  ->  out: [N,K] row-major bf16 (i.e. B transposed)
__global__ void k_transpose_cast(const float* __restrict__ in, unsigned short* __restrict__ out,
                                 int K, int N) {
    int i = blockIdx.x * blockDim.x + threadIdx.x;
    if (i < K * N) { int k = i / N, n = i % N; out[n * K + k] = f2bf(in[i]); }
}
__global__ void k_fill_u32(unsigned* __restrict__ p, unsigned v, int n) {
    int i = blockIdx.x * blockDim.x + threadIdx.x;
    if (i < n) p[i] = v;
}
__global__ void k_init_rst(float* __restrict__ rst, const float* __restrict__ bias, int n, int C) {
    int i = blockIdx.x * blockDim.x + threadIdx.x;
    if (i < n * C) rst[i] = bias[i % C];
}

// ---------- WMMA bf16 GEMM:  C[M,N] = A[M,K] * B[K,N],  Bt is B transposed [N,K] ----------
// block = 256 threads = 8 waves arranged 2(M) x 4(N); each wave computes a 32x32 tile
// (2 A-fragments x 2 B-fragments -> 4 WMMAs per k-step) => WG tile 64x128.
// Requires M%64==0, N%128==0, K%32==0.
union AB32 { unsigned u[8]; v16bf v; };

__device__ __forceinline__ void load_a_frag(const unsigned short* __restrict__ Ar,
                                            int k0, int g, AB32& f) {
#pragma unroll
    for (int i = 0; i < 8; ++i) {
        // A 16x32 bf16 fragment: VGPR i<4 -> K = 8g+2i(+1); i>=4 -> K = 16+8g+2(i-4)(+1)
        int ka = (i < 4) ? (8 * g + 2 * i) : (16 + 8 * g + 2 * (i - 4));
        f.u[i] = *(const unsigned*)(Ar + k0 + ka);
    }
}
__device__ __forceinline__ void load_b_frag(const unsigned short* __restrict__ Br,
                                            int k0, int g, AB32& f) {
#pragma unroll
    for (int i = 0; i < 8; ++i) {
        // B 32x16 bf16 fragment: VGPR i<4 -> K = 8g+2i(+1); i>=4 -> K = 16+8g+2(i-4)(+1), lane = column
        int kb = (i < 4) ? (8 * g + 2 * i) : (16 + 8 * g + 2 * (i - 4));
        f.u[i] = *(const unsigned*)(Br + k0 + kb);
    }
}

__global__ __launch_bounds__(TPB)
void k_wmma_gemm_bf16(const unsigned short* __restrict__ A,
                      const unsigned short* __restrict__ Bt,
                      float* __restrict__ C, int N, int K) {
    const int lane = threadIdx.x & 31;
    const int w    = threadIdx.x >> 5;
    const int m0   = blockIdx.x * 64  + (w & 1) * 32;
    const int n0   = blockIdx.y * 128 + (w >> 1) * 32;
    const int r15  = lane & 15;
    const int g    = lane >> 4;

    const unsigned short* Ar0 = A  + (size_t)(m0 + r15) * K;
    const unsigned short* Ar1 = Ar0 + (size_t)16 * K;
    const unsigned short* Br0 = Bt + (size_t)(n0 + r15) * K;
    const unsigned short* Br1 = Br0 + (size_t)16 * K;

    v8f c00 = {}, c01 = {}, c10 = {}, c11 = {};
    for (int k0 = 0; k0 < K; k0 += 32) {
        if (k0 + 32 < K) {
            __builtin_prefetch(Ar0 + k0 + 32, 0, 3);   // global_prefetch_b8
            __builtin_prefetch(Ar1 + k0 + 32, 0, 3);
            __builtin_prefetch(Br0 + k0 + 32, 0, 3);
            __builtin_prefetch(Br1 + k0 + 32, 0, 3);
        }
        AB32 a0, a1, b0, b1;
        load_a_frag(Ar0, k0, g, a0);
        load_a_frag(Ar1, k0, g, a1);
        load_b_frag(Br0, k0, g, b0);
        load_b_frag(Br1, k0, g, b1);
        c00 = __builtin_amdgcn_wmma_f32_16x16x32_bf16(false, a0.v, false, b0.v, (short)0, c00, false, false);
        c01 = __builtin_amdgcn_wmma_f32_16x16x32_bf16(false, a0.v, false, b1.v, (short)0, c01, false, false);
        c10 = __builtin_amdgcn_wmma_f32_16x16x32_bf16(false, a1.v, false, b0.v, (short)0, c10, false, false);
        c11 = __builtin_amdgcn_wmma_f32_16x16x32_bf16(false, a1.v, false, b1.v, (short)0, c11, false, false);
    }
    // C/D layout: VGPR r -> M = tile_m + 8*g + r, N = tile_n + (lane&15)
    float* C00 = C + (size_t)(m0 + 8 * g) * N + n0 + r15;
    float* C10 = C00 + (size_t)16 * N;
#pragma unroll
    for (int r = 0; r < 8; ++r) {
        C00[(size_t)r * N]      = c00[r];
        C00[(size_t)r * N + 16] = c01[r];
        C10[(size_t)r * N]      = c10[r];
        C10[(size_t)r * N + 16] = c11[r];
    }
}

// ---------- attention kernels (generic over H heads, D dims) ----------
__global__ void k_el_er(const float* __restrict__ feat, const float* __restrict__ al,
                        const float* __restrict__ ar, float* __restrict__ el,
                        float* __restrict__ er, int n, int H, int D) {
    int id = blockIdx.x * blockDim.x + threadIdx.x;
    if (id >= n * H) return;
    int node = id / H, h = id % H;
    const float* f  = feat + (size_t)node * H * D + (size_t)h * D;
    const float* aL = al + h * D;
    const float* aR = ar + h * D;
    float sl = 0.0f, sr = 0.0f;
    for (int d = 0; d < D; ++d) { float v = f[d]; sl += v * aL[d]; sr += v * aR[d]; }
    el[id] = sl; er[id] = sr;
}

__global__ void k_edge_max(const int* __restrict__ src, const int* __restrict__ dst,
                           const float* __restrict__ el, const float* __restrict__ er,
                           unsigned* __restrict__ menc, int E, int H) {
    int id = blockIdx.x * blockDim.x + threadIdx.x;
    if (id >= E * H) return;
    int e = id / H, h = id % H;
    int s = src[e], d = dst[e];
    float v = lrelu(el[s * H + h] + er[d * H + h]);
    atomicMax(&menc[d * H + h], fenc(v));
}

__global__ void k_edge_exp(const int* __restrict__ src, const int* __restrict__ dst,
                           const float* __restrict__ el, const float* __restrict__ er,
                           const unsigned* __restrict__ menc, float* __restrict__ eexp,
                           float* __restrict__ den, int E, int H) {
    int id = blockIdx.x * blockDim.x + threadIdx.x;
    if (id >= E * H) return;
    int e = id / H, h = id % H;
    int s = src[e], d = dst[e];
    float v = lrelu(el[s * H + h] + er[d * H + h]);
    float m = fdec(menc[d * H + h]);
    float ee = expf(v - m);
    eexp[id] = ee;
    atomicAdd(&den[d * H + h], ee);
}

__global__ void k_edge_aggr(const int* __restrict__ src, const int* __restrict__ dst,
                            const float* __restrict__ eexp, const float* __restrict__ den,
                            const float* __restrict__ feat, float* __restrict__ rst,
                            int E, int H, int D) {
    const int C = H * D;
    int id = blockIdx.x * blockDim.x + threadIdx.x;
    if (id >= E * C) return;
    int e = id / C, c = id % C, h = c / D;
    int s = src[e], d = dst[e];
    float alpha = eexp[(size_t)e * H + h] / den[d * H + h];
    atomicAdd(&rst[(size_t)d * C + c], alpha * feat[(size_t)s * C + c]);
}

// ---------- pooling ----------
__global__ void k_pool_count(const int* __restrict__ gid, float* __restrict__ cnt, int n) {
    int i = blockIdx.x * blockDim.x + threadIdx.x;
    if (i < n) atomicAdd(&cnt[gid[i]], 1.0f);
}
__global__ void k_pool_sum(const int* __restrict__ gid, const float* __restrict__ h,
                           float* __restrict__ out, int n, int C) {
    int i = blockIdx.x * blockDim.x + threadIdx.x;
    if (i >= n * C) return;
    int node = i / C, c = i % C;
    atomicAdd(&out[gid[node] * C + c], h[i]);
}
__global__ void k_pool_div(float* __restrict__ out, const float* __restrict__ cnt, int G, int C) {
    int i = blockIdx.x * blockDim.x + threadIdx.x;
    if (i < G * C) out[i] /= fmaxf(cnt[i / C], 1.0f);
}

// ---------- launch ----------
static inline size_t rup(size_t x) { return (x + 255) & ~(size_t)255; }
static inline int cdiv(long long a, long long b) { return (int)((a + b - 1) / b); }

extern "C" void kernel_launch(void* const* d_in, const int* in_sizes, int n_in,
                              void* d_out, int out_size, void* d_ws, size_t ws_size,
                              hipStream_t stream) {
    const float* x   = (const float*)d_in[0];
    const float* W1  = (const float*)d_in[1];
    const float* al1 = (const float*)d_in[2];
    const float* ar1 = (const float*)d_in[3];
    const float* b1  = (const float*)d_in[4];
    const float* W2  = (const float*)d_in[5];
    const float* al2 = (const float*)d_in[6];
    const float* ar2 = (const float*)d_in[7];
    const float* b2  = (const float*)d_in[8];
    const int* esrc  = (const int*)d_in[9];
    const int* edst  = (const int*)d_in[10];
    const int* gid   = (const int*)d_in[11];
    float* out = (float*)d_out;

    const int N = N_NODES, MP = M_PAD, E = N_EDGES, G = N_GRAPHS;

    char* ws = (char*)d_ws;
    size_t off = 0;
    auto alloc = [&](size_t b) { void* p = ws + off; off = rup(off + b); return p; };

    // ---- layer-1 temporaries (region A, reused by layer 2 afterwards) ----
    // A-side GEMM buffers are padded to MP rows; rows [N_NODES, MP) hold garbage that
    // is written by the GEMM but never read downstream.
    unsigned short* xb    = (unsigned short*)alloc((size_t)MP * 256 * 2);
    unsigned short* w1t   = (unsigned short*)alloc((size_t)512 * 256 * 2);
    float*          feat1 = (float*)alloc((size_t)MP * 512 * 4);
    float*          el1   = (float*)alloc((size_t)N * 8 * 4);
    float*          er1   = (float*)alloc((size_t)N * 8 * 4);
    unsigned*       m1    = (unsigned*)alloc((size_t)N * 8 * 4);
    float*          den1  = (float*)alloc((size_t)N * 8 * 4);
    float*          eexp1 = (float*)alloc((size_t)E * 8 * 4);
    // persistent across layers:
    float*          rst1  = (float*)alloc((size_t)N * 512 * 4);

    // ---- layer-2 temporaries alias region A (dead after layer-1 aggregation) ----
    off = 0;
    unsigned short* hb    = (unsigned short*)alloc((size_t)MP * 512 * 2);
    unsigned short* w2t   = (unsigned short*)alloc((size_t)128 * 512 * 2);
    float*          feat2 = (float*)alloc((size_t)MP * 128 * 4);
    float*          el2   = (float*)alloc((size_t)N * 4);
    float*          er2   = (float*)alloc((size_t)N * 4);
    unsigned*       m2    = (unsigned*)alloc((size_t)N * 4);
    float*          den2  = (float*)alloc((size_t)N * 4);
    float*          eexp2 = (float*)alloc((size_t)E * 4);
    float*          rst2  = (float*)alloc((size_t)N * 128 * 4);
    float*          cnt   = (float*)alloc((size_t)G * 4);

    // ================= layer 1 =================
    k_cast_bf16<<<cdiv((long long)N * 256, TPB), TPB, 0, stream>>>(x, xb, N * 256);
    k_transpose_cast<<<cdiv(256LL * 512, TPB), TPB, 0, stream>>>(W1, w1t, 256, 512);
    {
        dim3 grid(MP / 64, 512 / 128);
        k_wmma_gemm_bf16<<<grid, TPB, 0, stream>>>(xb, w1t, feat1, 512, 256);
    }
    k_el_er<<<cdiv((long long)N * 8, TPB), TPB, 0, stream>>>(feat1, al1, ar1, el1, er1, N, 8, 64);
    k_fill_u32<<<cdiv((long long)N * 8, TPB), TPB, 0, stream>>>(m1, 0u, N * 8);
    k_fill_u32<<<cdiv((long long)N * 8, TPB), TPB, 0, stream>>>((unsigned*)den1, 0u, N * 8);
    k_edge_max<<<cdiv((long long)E * 8, TPB), TPB, 0, stream>>>(esrc, edst, el1, er1, m1, E, 8);
    k_edge_exp<<<cdiv((long long)E * 8, TPB), TPB, 0, stream>>>(esrc, edst, el1, er1, m1, eexp1, den1, E, 8);
    k_init_rst<<<cdiv((long long)N * 512, TPB), TPB, 0, stream>>>(rst1, b1, N, 512);
    k_edge_aggr<<<cdiv((long long)E * 512, TPB), TPB, 0, stream>>>(esrc, edst, eexp1, den1, feat1, rst1, E, 8, 64);

    // ================= layer 2 =================
    k_cast_bf16<<<cdiv((long long)N * 512, TPB), TPB, 0, stream>>>(rst1, hb, N * 512);
    k_transpose_cast<<<cdiv(512LL * 128, TPB), TPB, 0, stream>>>(W2, w2t, 512, 128);
    {
        dim3 grid(MP / 64, 128 / 128);
        k_wmma_gemm_bf16<<<grid, TPB, 0, stream>>>(hb, w2t, feat2, 128, 512);
    }
    k_el_er<<<cdiv((long long)N, TPB), TPB, 0, stream>>>(feat2, al2, ar2, el2, er2, N, 1, 128);
    k_fill_u32<<<cdiv((long long)N, TPB), TPB, 0, stream>>>(m2, 0u, N);
    k_fill_u32<<<cdiv((long long)N, TPB), TPB, 0, stream>>>((unsigned*)den2, 0u, N);
    k_edge_max<<<cdiv((long long)E, TPB), TPB, 0, stream>>>(esrc, edst, el2, er2, m2, E, 1);
    k_edge_exp<<<cdiv((long long)E, TPB), TPB, 0, stream>>>(esrc, edst, el2, er2, m2, eexp2, den2, E, 1);
    k_init_rst<<<cdiv((long long)N * 128, TPB), TPB, 0, stream>>>(rst2, b2, N, 128);
    k_edge_aggr<<<cdiv((long long)E * 128, TPB), TPB, 0, stream>>>(esrc, edst, eexp2, den2, feat2, rst2, E, 1, 128);

    // ================= per-graph mean pooling =================
    k_fill_u32<<<cdiv((long long)G, TPB), TPB, 0, stream>>>((unsigned*)cnt, 0u, G);
    k_fill_u32<<<cdiv((long long)G * 128, TPB), TPB, 0, stream>>>((unsigned*)out, 0u, G * 128);
    k_pool_count<<<cdiv((long long)N, TPB), TPB, 0, stream>>>(gid, cnt, N);
    k_pool_sum<<<cdiv((long long)N * 128, TPB), TPB, 0, stream>>>(gid, rst2, out, N, 128);
    k_pool_div<<<cdiv((long long)G * 128, TPB), TPB, 0, stream>>>(out, cnt, G, 128);
}
